// MemoryModule_11441792876600
// MI455X (gfx1250) — matrix operand order repaired and checked by reading, not compiled, not run on hardware
//
#include <hip/hip_runtime.h>
#include <hip/hip_bf16.h>
#include <math.h>
#include <stdint.h>

typedef __bf16 bf16;
typedef __attribute__((ext_vector_type(16))) __bf16 v16bf;
typedef __attribute__((ext_vector_type(8)))  float  v8f;

#define B_ROWS 32768
#define N_MEM  2048
#define H_DIM  512

static __device__ __forceinline__ v8f wmma_bf16(v16bf a, v16bf b, v8f c) {
    // D = A(16x32 bf16) x B(32x16 bf16) + C(16x16 f32)
    return __builtin_amdgcn_wmma_f32_16x16x32_bf16(
        false, a, false, b, (short)0, c, false, false);
}

// v16bf fragment from two 16-byte pieces (elements 0..7 from p0, 8..15 from p1).
static __device__ __forceinline__ v16bf load_frag2(const bf16* p0, const bf16* p1) {
    union { v16bf v; uint4 u[2]; } r;
    r.u[0] = *(const uint4*)p0;
    r.u[1] = *(const uint4*)p1;
    return r.v;
}

static __device__ __forceinline__ v8f zero8() {
    v8f z;
#pragma unroll
    for (int i = 0; i < 8; ++i) z[i] = 0.0f;
    return z;
}

// ---- CDNA5 async global->LDS copy (ASYNCcnt path), 16B per lane ----
static __device__ __forceinline__ void async_b128(unsigned lds_off, const bf16* g) {
    asm volatile("global_load_async_to_lds_b128 %0, %1, off"
                 :: "v"(lds_off), "v"(g) : "memory");
}
static __device__ __forceinline__ void wait_async0() {
    asm volatile("s_wait_asynccnt 0x0" ::: "memory");
}

// ---------------- K0: normalize z rows -> bf16 ----------------
__global__ void k_norm_z(const float* __restrict__ z, bf16* __restrict__ zb) {
    const int lane = threadIdx.x & 31;
    const int wave = threadIdx.x >> 5;
    const int row  = blockIdx.x * 8 + wave;
    const float* zr = z + (size_t)row * H_DIM;
    float v[16];
    float ss = 0.0f;
#pragma unroll
    for (int i = 0; i < 16; ++i) { v[i] = zr[lane + i * 32]; ss += v[i] * v[i]; }
#pragma unroll
    for (int off = 1; off < 32; off <<= 1) ss += __shfl_xor(ss, off, 32);
    const float sc = 1.0f / fmaxf(sqrtf(ss), 1e-12f);
    bf16* zo = zb + (size_t)row * H_DIM;
#pragma unroll
    for (int i = 0; i < 16; ++i) zo[lane + i * 32] = (bf16)(v[i] * sc);
}

// ------- K1: memory prep: normalized bf16 m_n, raw transposed bf16 memT -------
__global__ void k_prep_mem(const float* __restrict__ mem,
                           bf16* __restrict__ mn, bf16* __restrict__ mT) {
    const int lane = threadIdx.x & 31;
    const int wave = threadIdx.x >> 5;
    const int row  = blockIdx.x * 8 + wave;
    const float* mr = mem + (size_t)row * H_DIM;
    float v[16];
    float ss = 0.0f;
#pragma unroll
    for (int i = 0; i < 16; ++i) { v[i] = mr[lane + i * 32]; ss += v[i] * v[i]; }
#pragma unroll
    for (int off = 1; off < 32; off <<= 1) ss += __shfl_xor(ss, off, 32);
    const float sc = 1.0f / fmaxf(sqrtf(ss), 1e-12f);
    bf16* mo = mn + (size_t)row * H_DIM;
#pragma unroll
    for (int i = 0; i < 16; ++i) {
        const int h = lane + i * 32;
        mo[h] = (bf16)(v[i] * sc);
        mT[(size_t)h * N_MEM + row] = (bf16)v[i];   // raw (not normalized)
    }
}

// ---------------- Kzero ----------------
__global__ void k_zero(float* __restrict__ p, int n) {
    const int i = blockIdx.x * 256 + threadIdx.x;
    if (i < n) p[i] = 0.0f;
}

// LDS layout (bytes):
//   [0, 34816)      : union { stage1[2] (2 x 128 rows x 40 bf16 = 2 x 10240B) ;
//                             pbuf (8 waves x 16 x 136 bf16 = 34816B) }
//   [34816, 52224)  : stage2[2] (2 x 32 rows x 136 bf16 = 2 x 8704B)
#define S1_STRIDE 40    // bf16 elements per staged m_n row (bank-conflict pad)
#define S2_STRIDE 136   // bf16 elements per staged memT row
#define PB_STRIDE 136
#define SMEM_BYTES 52224

// stage 8KB slice m_n[c0..c0+127][kk..kk+31] -> s1buf (512 x 16B async ops)
static __device__ __forceinline__ void stage1_issue(bf16* s1buf, const bf16* mn,
                                                    int c0, int kk, int tid) {
    const unsigned base = (unsigned)(uintptr_t)s1buf;
#pragma unroll
    for (int i = 0; i < 2; ++i) {
        const int o = tid + i * 256;
        const int r = o >> 2, c = o & 3;
        async_b128(base + (unsigned)(r * (S1_STRIDE * 2) + c * 16),
                   mn + (size_t)(c0 + r) * H_DIM + kk + c * 8);
    }
}

// stage 8KB slice memT[h0..h0+31][c0..c0+127] -> s2buf
static __device__ __forceinline__ void stage2_issue(bf16* s2buf, const bf16* mT,
                                                    int c0, int h0, int tid) {
    const unsigned base = (unsigned)(uintptr_t)s2buf;
#pragma unroll
    for (int i = 0; i < 2; ++i) {
        const int o = tid + i * 256;
        const int row = o >> 4, seg = o & 15;
        async_b128(base + (unsigned)(row * (S2_STRIDE * 2) + seg * 16),
                   mT + (size_t)(h0 + row) * N_MEM + c0 + seg * 8);
    }
}

// ---------------- K2: fused similarity/softmax/read + update scatter ----------------
__global__ __launch_bounds__(256) void k_fused(
    const bf16* __restrict__ z_bf, const bf16* __restrict__ m_n,
    const bf16* __restrict__ memT, const float* __restrict__ z_raw,
    float* __restrict__ zhat, float* __restrict__ upd_num,
    float* __restrict__ upd_den, float* __restrict__ upd_cnt) {

    const int tid  = threadIdx.x;
    const int lane = tid & 31;
    const int wave = tid >> 5;
    const int half = lane >> 4;
    const int ln   = lane & 15;
    const int rowBase = blockIdx.x * 128 + wave * 16;

    __shared__ __align__(16) char smem[SMEM_BYTES];
    bf16* s1[2] = { (bf16*)smem, (bf16*)(smem + 10240) };
    bf16* pb    = (bf16*)smem + wave * 16 * PB_STRIDE;      // aliases stage1 (time-disjoint)
    bf16* s2[2] = { (bf16*)(smem + 34816), (bf16*)(smem + 34816 + 8704) };

    const bf16* zrow = z_bf + (size_t)(rowBase + ln) * H_DIM + half * 16;

    v8f acc2[32];
#pragma unroll
    for (int i = 0; i < 32; ++i) acc2[i] = zero8();

    float rowsum[8], rowmax[8];
    int   argn[8];
#pragma unroll
    for (int j = 0; j < 8; ++j) { rowsum[j] = 0.0f; rowmax[j] = -1e30f; argn[j] = 0; }

#pragma unroll 1
    for (int c0 = 0; c0 < N_MEM; c0 += 128) {
        // ================= GEMM1: sim(16x128) = z_n @ m_n^T =================
        v8f simf[8];
#pragma unroll
        for (int t = 0; t < 8; ++t) simf[t] = zero8();

        // Launder a scalar offset (SGPR) so the per-k A loads are re-issued per
        // chunk (LICM can't pin 128 VGPRs -> no scratch spills), while the GEP
        // base stays the global kernarg pointer -> global_load (LOADcnt only).
        unsigned long long zoff = 0;
        asm volatile("" : "+s"(zoff));
        const bf16* zr = zrow + zoff;

        stage1_issue(s1[0], m_n, c0, 0, tid);     // prologue: stage k=0
        v16bf Acur = load_frag2(zr, zr + 8);      // A(k=0)
#pragma unroll
        for (int k = 0; k < 16; ++k) {
            wait_async0();
            __syncthreads();                       // buf[k&1] ready for all waves
            if (k < 15) stage1_issue(s1[(k + 1) & 1], m_n, c0, (k + 1) * 32, tid);
            v16bf Anext = Acur;
            if (k < 15)
                Anext = load_frag2(zr + (k + 1) * 32, zr + (k + 1) * 32 + 8);
            const bf16* sb = s1[k & 1];
            const bf16* bp0 = sb + ln * S1_STRIDE + half * 16;
            v16bf Bcur = load_frag2(bp0, bp0 + 8);
#pragma unroll
            for (int t = 0; t < 8; ++t) {
                v16bf Bnext = Bcur;
                if (t < 7) {
                    const bf16* bp = sb + ((t + 1) * 16 + ln) * S1_STRIDE + half * 16;
                    Bnext = load_frag2(bp, bp + 8);
                }
                simf[t] = wmma_bf16(Acur, Bcur, simf[t]);
                Bcur = Bnext;
            }
            Acur = Anext;
        }
        wait_async0();
        __syncthreads();   // all waves done reading stage1 before pbuf overwrite

        // ---- epilogue: exp, running row stats, stash p (bf16) in LDS ----
#pragma unroll
        for (int t = 0; t < 8; ++t) {
            const int col = c0 + t * 16 + ln;
#pragma unroll
            for (int j = 0; j < 8; ++j) {
                const float s = simf[t][j];
                const float e = __expf(s);         // safe: |s| <= 1
                rowsum[j] += e;
                if (s > rowmax[j]) { rowmax[j] = s; argn[j] = col; }
                pb[(j + 8 * half) * PB_STRIDE + t * 16 + ln] = (bf16)e;
            }
        }

        // ---- reload p in A-layout (same-wave LDS: in order) ----
        v16bf A2[4];
#pragma unroll
        for (int k2 = 0; k2 < 4; ++k2) {
            const bf16* pp = pb + ln * PB_STRIDE + k2 * 32 + half * 16;
            A2[k2] = load_frag2(pp, pp + 8);
        }

        // ================= GEMM2: acc2 += p(16x128) @ memory(128xH) =================
        stage2_issue(s2[0], memT, c0, 0, tid);    // prologue: h0=0
#pragma unroll
        for (int hs = 0; hs < 16; ++hs) {
            wait_async0();
            __syncthreads();                       // also fences pbuf vs next stage1
            if (hs < 15) stage2_issue(s2[(hs + 1) & 1], memT, c0, (hs + 1) * 32, tid);
            const bf16* sb = s2[hs & 1];
            const bf16* bq0 = sb + ln * S2_STRIDE + half * 16;
            v16bf Bcur = load_frag2(bq0, bq0 + 8);
#pragma unroll
            for (int q = 0; q < 8; ++q) {          // q = (ti<<2)|k2
                v16bf Bnext = Bcur;
                if (q < 7) {
                    const int qn = q + 1;
                    const bf16* bp = sb + ((qn >> 2) * 16 + ln) * S2_STRIDE
                                        + (qn & 3) * 32 + half * 16;
                    Bnext = load_frag2(bp, bp + 8);
                }
                acc2[hs * 2 + (q >> 2)] =
                    wmma_bf16(A2[q & 3], Bcur, acc2[hs * 2 + (q >> 2)]);
                Bcur = Bnext;
            }
        }
    }

    // ---- reduce row stats across the 16 lanes of each half ----
#pragma unroll
    for (int j = 0; j < 8; ++j) {
        float rs = rowsum[j];
#pragma unroll
        for (int off = 1; off < 16; off <<= 1) rs += __shfl_xor(rs, off, 32);
        rowsum[j] = rs;
        float mv = rowmax[j]; int an = argn[j];
#pragma unroll
        for (int off = 1; off < 16; off <<= 1) {
            const float omv = __shfl_xor(mv, off, 32);
            const int   oan = __shfl_xor(an, off, 32);
            if (omv > mv) { mv = omv; an = oan; }
        }
        rowmax[j] = __shfl(mv, half * 16, 32);   // broadcast for consistent ties
        argn[j]   = __shfl(an, half * 16, 32);
    }

    // ---- z_hat = acc2 / rowsum ----
#pragma unroll
    for (int ht = 0; ht < 32; ++ht) {
#pragma unroll
        for (int j = 0; j < 8; ++j) {
            zhat[(size_t)(rowBase + j + 8 * half) * H_DIM + ht * 16 + ln] =
                acc2[ht][j] / rowsum[j];
        }
    }

    // ---- update scatter: argmax slot accumulates exp(sim)*z[row,:] ----
#pragma unroll
    for (int j = 0; j < 8; ++j) {
        const int   row = rowBase + j + 8 * half;
        const int   ic  = argn[j];
        const float ev  = __expf(rowmax[j]);
        if (ln == 0) {
            atomicAdd(&upd_den[ic], ev);
            atomicAdd(&upd_cnt[ic], 1.0f);
        }
        const float* zr = z_raw + (size_t)row * H_DIM;
        float* un = upd_num + (size_t)ic * H_DIM;
        for (int hh = ln; hh < H_DIM; hh += 16) atomicAdd(&un[hh], ev * zr[hh]);
    }
}

// ---------------- K3: finalize memory ----------------
__global__ void k_final_mem(const float* __restrict__ mem, const float* __restrict__ num,
                            const float* __restrict__ den, const float* __restrict__ cnt,
                            float* __restrict__ out) {
    const int lane = threadIdx.x & 31;
    const int wave = threadIdx.x >> 5;
    const int row  = blockIdx.x * 8 + wave;
    const float* mr = mem + (size_t)row * H_DIM;
    float* orow = out + (size_t)row * H_DIM;
    if (cnt[row] > 0.0f) {
        const float d = fmaxf(den[row], 1e-30f);
        const float* nr = num + (size_t)row * H_DIM;
        float v[16];
        float ss = 0.0f;
#pragma unroll
        for (int i = 0; i < 16; ++i) {
            const int h = lane + i * 32;
            v[i] = mr[h] + nr[h] / d;
            ss += v[i] * v[i];
        }
#pragma unroll
        for (int off = 1; off < 32; off <<= 1) ss += __shfl_xor(ss, off, 32);
        const float sc = 1.0f / fmaxf(sqrtf(ss), 1e-12f);
#pragma unroll
        for (int i = 0; i < 16; ++i) orow[lane + i * 32] = v[i] * sc;
    } else {
#pragma unroll
        for (int i = 0; i < 16; ++i) orow[lane + i * 32] = mr[lane + i * 32];
    }
}

extern "C" void kernel_launch(void* const* d_in, const int* in_sizes, int n_in,
                              void* d_out, int out_size, void* d_ws, size_t ws_size,
                              hipStream_t stream) {
    const float* z   = (const float*)d_in[0];   // (B,H) f32
    const float* mem = (const float*)d_in[1];   // (N,H) f32
    float* zhat    = (float*)d_out;                          // (B,H)
    float* new_mem = (float*)d_out + (size_t)B_ROWS * H_DIM; // (N,H)

    char* ws = (char*)d_ws;
    size_t off = 0;
    bf16*  z_bf    = (bf16*)(ws + off); off += (size_t)B_ROWS * H_DIM * sizeof(bf16);
    bf16*  m_n     = (bf16*)(ws + off); off += (size_t)N_MEM * H_DIM * sizeof(bf16);
    bf16*  memT    = (bf16*)(ws + off); off += (size_t)H_DIM * N_MEM * sizeof(bf16);
    float* upd_num = (float*)(ws + off); off += (size_t)N_MEM * H_DIM * sizeof(float);
    float* upd_den = (float*)(ws + off); off += (size_t)N_MEM * sizeof(float);
    float* upd_cnt = (float*)(ws + off); off += (size_t)N_MEM * sizeof(float);
    const int nz = N_MEM * H_DIM + 2 * N_MEM;   // contiguous -> single zero pass

    k_norm_z   <<<B_ROWS / 8, 256, 0, stream>>>(z, z_bf);
    k_prep_mem <<<N_MEM / 8, 256, 0, stream>>>(mem, m_n, memT);
    k_zero     <<<(nz + 255) / 256, 256, 0, stream>>>(upd_num, nz);
    k_fused    <<<B_ROWS / 128, 256, 0, stream>>>(z_bf, m_n, memT, z,
                                                  zhat, upd_num, upd_den, upd_cnt);
    k_final_mem<<<N_MEM / 8, 256, 0, stream>>>(mem, upd_num, upd_den, upd_cnt, new_mem);
}
